// Bidirect_LSTM_46651934769704
// MI455X (gfx1250) — matrix operand order repaired and checked by reading, not compiled, not run on hardware
//
#include <hip/hip_runtime.h>

typedef __attribute__((ext_vector_type(16))) _Float16 v16h;
typedef __attribute__((ext_vector_type(8)))  _Float16 v8h;
typedef __attribute__((ext_vector_type(8)))  float    v8f;

#define EMBED 300
#define EPAD  320      // EMBED padded to multiple of 32 (WMMA K-step)
#define HID   1024
#define GATES 4096
#define BATCH 64
#define SEQ   128

#define NBLK  64       // persistent blocks: one 16-wide hidden column each
#define LDSLD 1352     // LDS row stride in halves: 2704B = 676 dw; r*36 mod 64
                       // distinct for r=0..15 -> conflict-free ds_load_b128
#define KTOT  (EPAD + HID)   // 1344: unified K (x part then h part)

#define WMMA_F16(a, b, c) \
    __builtin_amdgcn_wmma_f32_16x16x32_f16(false, (a), false, (b), (short)0, (c), false, false)

// Branch-free fast activations (saturate correctly at large |x|, NaN-propagating)
__device__ __forceinline__ float sigmoidf_(float x) {
    return 1.0f / (1.0f + __expf(-x));
}
__device__ __forceinline__ float tanhf_(float x) {
    return 1.0f - 2.0f / (__expf(2.0f * x) + 1.0f);
}

// ---------------------------------------------------------------------------
// Prep kernels (cheap, once per kernel_launch)
// ---------------------------------------------------------------------------

// embR[t][b][e] = f16(embed_table[inputs[b][SEQ-1-t]][e]), zero-padded e>=300.
__global__ void k_gather_embeds(const float* __restrict__ table,
                                const int* __restrict__ idx,
                                _Float16* __restrict__ embR) {
    int i = blockIdx.x * blockDim.x + threadIdx.x;
    if (i >= SEQ * BATCH * EPAD) return;
    int e  = i % EPAD;
    int rb = i / EPAD;
    int b  = rb % BATCH;
    int t  = rb / BATCH;                       // reversed time index
    int tok = idx[b * SEQ + (SEQ - 1 - t)];
    float v = (e < EMBED) ? table[(size_t)tok * EMBED + e] : 0.0f;
    embR[i] = (_Float16)v;
}

__global__ void k_convert_pad(const float* __restrict__ src,
                              _Float16* __restrict__ dst,
                              int rows, int src_cols, int dst_cols) {
    int i = blockIdx.x * blockDim.x + threadIdx.x;
    if (i >= rows * dst_cols) return;
    int c = i % dst_cols;
    int r = i / dst_cols;
    dst[i] = (_Float16)((c < src_cols) ? src[(size_t)r * src_cols + c] : 0.0f);
}

__global__ void k_bias_sum(const float* __restrict__ a,
                           const float* __restrict__ b,
                           float* __restrict__ o, int n) {
    int i = blockIdx.x * blockDim.x + threadIdx.x;
    if (i < n) o[i] = a[i] + b[i];
}

// zero h0 and the grid-barrier counter (ws is poisoned 0xAA by the harness)
__global__ void k_init_state(_Float16* __restrict__ h, unsigned* __restrict__ bar) {
    int i = blockIdx.x * blockDim.x + threadIdx.x;
    if (i < BATCH * HID) h[i] = (_Float16)0.0f;
    if (i == 0) *bar = 0u;
}

// ---------------------------------------------------------------------------
// WMMA fragment loaders (CDNA5 16x16x32 f16 layouts, wave32)
// ---------------------------------------------------------------------------

// A: 16x32 (MxK) from row-major global, ld halves per row.
// lane&15 = M row; lanes 0-15 hold K=kc+0..15, lanes 16-31 K=kc+16..31.
__device__ __forceinline__ v16h load_a_frag(const _Float16* __restrict__ base,
                                            int ld, int m0, int kc, int lane) {
    int m  = m0 + (lane & 15);
    int hi = lane >> 4;
    const _Float16* p = base + (size_t)m * ld + kc + hi * 16;
    v8h lo  = *(const v8h*)(p);        // K = kc + hi*16 + 0..7
    v8h hi8 = *(const v8h*)(p + 8);    // K = kc + hi*16 + 8..15
    v16h a;
#pragma unroll
    for (int i = 0; i < 8; ++i) { a[i] = lo[i]; a[i + 8] = hi8[i]; }
    return a;
}

// B: 32x16 (KxN), B[k][n] = W[n][k]; weight rows resident in LDS.
// lane&15 = N (a weight row); lanes 0-15 hold K=kc+0..15, lanes 16-31 K=kc+16..31.
__device__ __forceinline__ v16h load_b_lds(const _Float16* lds, int row, int kc, int hi) {
    const _Float16* p = lds + (size_t)row * LDSLD + kc + hi * 16;
    v8h lo  = *(const v8h*)(p);        // ds_load_b128
    v8h hi8 = *(const v8h*)(p + 8);
    v16h b;
#pragma unroll
    for (int i = 0; i < 8; ++i) { b[i] = lo[i]; b[i + 8] = hi8[i]; }
    return b;
}

// B directly from global (forward one-shot kernel)
__device__ __forceinline__ v16h load_b_glb(const _Float16* __restrict__ w,
                                           int ld, int nrow, int kc, int lane) {
    int hi = lane >> 4;
    const _Float16* p = w + (size_t)nrow * ld + kc + hi * 16;
    v8h lo  = *(const v8h*)(p);
    v8h hi8 = *(const v8h*)(p + 8);
    v16h b;
#pragma unroll
    for (int i = 0; i < 8; ++i) { b[i] = lo[i]; b[i + 8] = hi8[i]; }
    return b;
}

// ---------------------------------------------------------------------------
// Forward direction (reference bug: h=c=0, so h@Whh==0):
//   h_fwd = sigmoid(o) * tanh( sigmoid(i) * tanh(g) ),  gates = x_last @ Wih_f.T + b
// One wave per 16x16 tile of [64,1024]; writes d_out[:, 0:1024].
// ---------------------------------------------------------------------------
__global__ void k_cell_fwd(const _Float16* __restrict__ X,     // [BATCH][EPAD]
                           const _Float16* __restrict__ Wih,   // [GATES][EPAD]
                           const float* __restrict__ bias,     // [GATES]
                           float* __restrict__ out32) {        // stride 2*HID
    int lane = threadIdx.x & 31;
    int wid  = (blockIdx.x * blockDim.x + threadIdx.x) >> 5;  // 0..255
    int m0   = (wid >> 6) * 16;
    int n0   = (wid & 63) * 16;
    int nrow = n0 + (lane & 15);

    v8f ai = {}, ag = {}, ao = {};
#pragma unroll
    for (int kc = 0; kc < EPAD; kc += 32) {
        v16h a  = load_a_frag(X, EPAD, m0, kc, lane);
        v16h bi = load_b_glb(Wih, EPAD, nrow,           kc, lane);
        v16h bg = load_b_glb(Wih, EPAD, nrow + 2 * HID, kc, lane);
        v16h bo = load_b_glb(Wih, EPAD, nrow + 3 * HID, kc, lane);
        ai = WMMA_F16(a, bi, ai);
        ag = WMMA_F16(a, bg, ag);
        ao = WMMA_F16(a, bo, ao);
    }
    float bi_s = bias[nrow];
    float bg_s = bias[nrow + 2 * HID];
    float bo_s = bias[nrow + 3 * HID];
    int col   = n0 + (lane & 15);
    int mbase = m0 + ((lane >> 4) << 3);
#pragma unroll
    for (int r = 0; r < 8; ++r) {
        float gi = sigmoidf_(ai[r] + bi_s);
        float gg = tanhf_(ag[r] + bg_s);
        float go = sigmoidf_(ao[r] + bo_s);
        out32[(size_t)(mbase + r) * (2 * HID) + col] = go * tanhf_(gi * gg);
    }
}

// ---------------------------------------------------------------------------
// Persistent backward recurrence.
//   64 blocks x 128 threads (4 waves). Block owns one 16-wide hidden column;
//   wave w owns batch tile m0 = 16*w. Weight rows [Wih | Whh] for gates i,g,o
//   of this column (48 rows x 1344 halves, ~130 KB) are staged to LDS ONCE,
//   then the block loops all 128 timesteps with a device-wide generation
//   barrier. h is double-buffered f16 in global; c==0 forever (reference
//   bug), so h = sigmoid(o) * tanh( sigmoid(i) * tanh(g) ).
// ---------------------------------------------------------------------------
__global__ void k_lstm_persistent(const _Float16* __restrict__ embR,  // [SEQ][BATCH][EPAD]
                                  const _Float16* __restrict__ Wih,   // [GATES][EPAD]
                                  const _Float16* __restrict__ Whh,   // [GATES][HID]
                                  const float* __restrict__ bias,     // [GATES]
                                  _Float16* __restrict__ h0,          // [BATCH][HID]
                                  _Float16* __restrict__ h1,          // [BATCH][HID]
                                  float* __restrict__ out32,          // d_out + HID, stride 2*HID
                                  unsigned* __restrict__ bar) {
    extern __shared__ _Float16 lds[];           // 48 rows x LDSLD halves (~130 KB)
    const int tid  = threadIdx.x;
    const int lane = tid & 31;
    const int wave = tid >> 5;                  // 0..3 -> batch tile
    const int n0   = blockIdx.x * 16;           // hidden column base

    // ---- stage weights into LDS: row = gate*16 + r ; cols [0,320)=Wih, [320,1344)=Whh
    for (int idx = tid; idx < 48 * (KTOT / 8); idx += blockDim.x) {
        int row = idx / (KTOT / 8);             // 0..47
        int g8  = idx % (KTOT / 8);             // 8-half group within the 1344 row
        int gate = row >> 4;                    // 0:i 1:g 2:o
        int r    = row & 15;
        int wrow = n0 + r + (gate == 0 ? 0 : (gate == 1 ? 2 * HID : 3 * HID));
        v8h v;
        if (g8 < EPAD / 8) v = *(const v8h*)(Wih + (size_t)wrow * EPAD + g8 * 8);
        else               v = *(const v8h*)(Whh + (size_t)wrow * HID + (g8 - EPAD / 8) * 8);
        *(v8h*)(lds + (size_t)row * LDSLD + g8 * 8) = v;
    }
    __syncthreads();

    const int m0    = wave * 16;
    const int colr  = lane & 15;
    const int hi    = lane >> 4;
    const int col   = n0 + colr;
    const int mbase = m0 + (hi << 3);
    const float bi_s = bias[col];
    const float bg_s = bias[col + 2 * HID];
    const float bo_s = bias[col + 3 * HID];

    _Float16* hbuf[2] = { h0, h1 };

    for (int t = 0; t < SEQ; ++t) {
        const _Float16* X     = embR + (size_t)t * BATCH * EPAD;
        const _Float16* hprev = hbuf[t & 1];
        _Float16*       hnext = hbuf[(t + 1) & 1];

        v8f ai = {}, ag = {}, ao = {};

        // ---- x part: K = 0..319 (A from global, double-buffered; B from LDS)
        v16h a_cur = load_a_frag(X, EPAD, m0, 0, lane);
#pragma unroll
        for (int c = 0; c < EPAD / 32; ++c) {
            int kl = c * 32;
            v16h a_nxt = (c + 1 < EPAD / 32)
                             ? load_a_frag(X, EPAD, m0, kl + 32, lane)
                             : load_a_frag(hprev, HID, m0, 0, lane);  // first h chunk
            v16h bi = load_b_lds(lds,  0 + colr, kl, hi);
            v16h bg = load_b_lds(lds, 16 + colr, kl, hi);
            v16h bo = load_b_lds(lds, 32 + colr, kl, hi);
            ai = WMMA_F16(a_cur, bi, ai);
            ag = WMMA_F16(a_cur, bg, ag);
            ao = WMMA_F16(a_cur, bo, ao);
            a_cur = a_nxt;
        }
        // ---- h part: K = 320..1343 (a_cur already holds h chunk 0)
#pragma unroll 4
        for (int c = 0; c < HID / 32; ++c) {
            int kl = EPAD + c * 32;
            v16h a_nxt = a_cur;
            if (c + 1 < HID / 32)
                a_nxt = load_a_frag(hprev, HID, m0, (c + 1) * 32, lane);
            v16h bi = load_b_lds(lds,  0 + colr, kl, hi);
            v16h bg = load_b_lds(lds, 16 + colr, kl, hi);
            v16h bo = load_b_lds(lds, 32 + colr, kl, hi);
            ai = WMMA_F16(a_cur, bi, ai);
            ag = WMMA_F16(a_cur, bg, ag);
            ao = WMMA_F16(a_cur, bo, ao);
            a_cur = a_nxt;
        }

        // ---- activations + h writeback (C/D layout: row r + 8*hi, col lane&15)
#pragma unroll
        for (int r = 0; r < 8; ++r) {
            float gi = sigmoidf_(ai[r] + bi_s);
            float gg = tanhf_(ag[r] + bg_s);
            float go = sigmoidf_(ao[r] + bo_s);
            float h  = go * tanhf_(gi * gg);    // c == 0 always (reference bug)
            hnext[(size_t)(mbase + r) * HID + col] = (_Float16)h;
            if (t == SEQ - 1)
                out32[(size_t)(mbase + r) * (2 * HID) + col] = h;
        }

        // ---- device-wide generation barrier (monotonic counter, graph-safe)
        __threadfence();                        // h stores visible at agent scope
        __syncthreads();                        // whole block done storing
        if (tid == 0) {
            __hip_atomic_fetch_add(bar, 1u, __ATOMIC_RELEASE, __HIP_MEMORY_SCOPE_AGENT);
            unsigned target = (unsigned)(t + 1) * NBLK;
            while (__hip_atomic_load(bar, __ATOMIC_ACQUIRE, __HIP_MEMORY_SCOPE_AGENT) < target)
                __builtin_amdgcn_s_sleep(1);
        }
        __syncthreads();                        // block waits on tid 0
    }
}

// ---------------------------------------------------------------------------
extern "C" void kernel_launch(void* const* d_in, const int* in_sizes, int n_in,
                              void* d_out, int out_size, void* d_ws, size_t ws_size,
                              hipStream_t stream) {
    const float* table = (const float*)d_in[0];
    const float* Wih_f = (const float*)d_in[1];
    // d_in[2] = Whh_f: multiplied by all-zero h in the reference -> unused
    const float* bih_f = (const float*)d_in[3];
    const float* bhh_f = (const float*)d_in[4];
    const float* Wih_b = (const float*)d_in[5];
    const float* Whh_b = (const float*)d_in[6];
    const float* bih_b = (const float*)d_in[7];
    const float* bhh_b = (const float*)d_in[8];
    const int*   toks  = (const int*)d_in[9];
    float* out = (float*)d_out;

    // Workspace carve-out (256B-aligned; ~19 MB total)
    char* ws = (char*)d_ws;
    size_t off = 0;
    auto carve = [&](size_t bytes) {
        void* p = ws + off;
        off = (off + bytes + 255) & ~(size_t)255;
        return p;
    };
    _Float16* embR   = (_Float16*)carve((size_t)SEQ * BATCH * EPAD * 2);
    _Float16* wihF16 = (_Float16*)carve((size_t)GATES * EPAD * 2);
    _Float16* wihB16 = (_Float16*)carve((size_t)GATES * EPAD * 2);
    _Float16* whhB16 = (_Float16*)carve((size_t)GATES * HID * 2);
    float*    biasF  = (float*)carve((size_t)GATES * 4);
    float*    biasB  = (float*)carve((size_t)GATES * 4);
    _Float16* h0     = (_Float16*)carve((size_t)BATCH * HID * 2);
    _Float16* h1     = (_Float16*)carve((size_t)BATCH * HID * 2);
    unsigned* bar    = (unsigned*)carve(256);

    // --- prep ---
    int n1 = SEQ * BATCH * EPAD;
    k_gather_embeds<<<(n1 + 255) / 256, 256, 0, stream>>>(table, toks, embR);
    int n2 = GATES * EPAD;
    k_convert_pad<<<(n2 + 255) / 256, 256, 0, stream>>>(Wih_f, wihF16, GATES, EMBED, EPAD);
    k_convert_pad<<<(n2 + 255) / 256, 256, 0, stream>>>(Wih_b, wihB16, GATES, EMBED, EPAD);
    int n3 = GATES * HID;
    k_convert_pad<<<(n3 + 255) / 256, 256, 0, stream>>>(Whh_b, whhB16, GATES, HID, HID);
    k_bias_sum<<<(GATES + 255) / 256, 256, 0, stream>>>(bih_f, bhh_f, biasF, GATES);
    k_bias_sum<<<(GATES + 255) / 256, 256, 0, stream>>>(bih_b, bhh_b, biasB, GATES);
    k_init_state<<<(BATCH * HID + 255) / 256, 256, 0, stream>>>(h0, bar);

    // --- forward direction: single cell on last token (embR slab t=0) ---
    k_cell_fwd<<<32, 256, 0, stream>>>(embR, wihF16, biasF, out);

    // --- backward recurrence: one persistent kernel, 128 internal steps ---
    size_t ldsBytes = (size_t)48 * LDSLD * sizeof(_Float16);   // ~130 KB
    k_lstm_persistent<<<NBLK, 128, ldsBytes, stream>>>(embR, wihB16, whhB16, biasB,
                                                       h0, h1, out + HID, bar);
}
